// F1AeroNetV2_84232898609315
// MI455X (gfx1250) — matrix operand order repaired and checked by reading, not compiled, not run on hardware
//
#include <hip/hip_runtime.h>
#include <math.h>

#define N_FINE   40000
#define N_COARSE 4000
#define E_FINE   640000
#define E_COARSE 64000
#define IN_CH    4
#define SPROJ    64
#define HEAD_H   128
#define NNL      7
#define PI2f     6.283185307179586f

typedef __bf16 bf16;
typedef __attribute__((ext_vector_type(16))) __bf16 v16bf;
typedef __attribute__((ext_vector_type(8)))  __bf16 v8bf;
typedef __attribute__((ext_vector_type(8)))  float  v8f;
typedef __attribute__((ext_vector_type(4)))  float  v4f;

__device__ __forceinline__ bf16 f2bf(float f) { return (bf16)f; }

__device__ __forceinline__ void put4(v16bf& a, int base, v4f f) {
  a[base + 0] = (bf16)f.x; a[base + 1] = (bf16)f.y;
  a[base + 2] = (bf16)f.z; a[base + 3] = (bf16)f.w;
}

__device__ __forceinline__ v8f wmma_bf16(v16bf a, v16bf b, v8f c) {
  return __builtin_amdgcn_wmma_f32_16x16x32_bf16(false, a, false, b, (short)0, c, false, false);
}

template <bool NTA>
__device__ __forceinline__ v4f ld4(const float* p) {
  if (NTA) return __builtin_nontemporal_load((const v4f*)p);   // TH=NT one-shot stream
  return *(const v4f*)p;
}

__device__ __forceinline__ void store_tile(float* __restrict__ C, v8f acc,
                                           const float* __restrict__ bias,
                                           int m0, int ccol, int N, int hi16, int relu) {
  float bb = bias ? bias[ccol] : 0.f;
#pragma unroll
  for (int r = 0; r < 8; ++r) {
    int row = m0 + r + hi16 * 8;
    float v = acc[r] + bb;
    if (relu) v = fmaxf(v, 0.f);
    C[(long)row * N + ccol] = v;
  }
}

// ---------------- small VALU kernels ----------------

__global__ void k_linear_small(const float* __restrict__ A, const float* __restrict__ W,
                               const float* __restrict__ b, float* __restrict__ C,
                               int M, int K, int N) {
  long t = (long)blockIdx.x * blockDim.x + threadIdx.x;
  if (t >= (long)M * N) return;
  int m = (int)(t / N), n = (int)(t % N);
  float acc = b ? b[n] : 0.f;
  for (int k = 0; k < K; ++k) acc += A[(long)m * K + k] * W[(long)k * N + n];
  C[t] = acc;
}

__global__ void k_gather_rows(const float* __restrict__ h, const int* __restrict__ idx,
                              float* __restrict__ out, int n, int d) {
  long t = (long)blockIdx.x * blockDim.x + threadIdx.x;
  if (t >= (long)n * d) return;
  int r = (int)(t / d), c = (int)(t % d);
  out[t] = h[(long)idx[r] * d + c];
}

__global__ void k_edge_deg(const int* __restrict__ dst, float* __restrict__ deg, int E) {
  int t = blockIdx.x * blockDim.x + threadIdx.x;
  if (t < E) atomicAdd(&deg[dst[t]], 1.0f);
}

__global__ void k_finalize_conv(float* __restrict__ out, const float* __restrict__ agg,
                                const float* __restrict__ deg, int n, int d) {
  long t = (long)blockIdx.x * blockDim.x + threadIdx.x;
  if (t >= (long)n * d) return;
  int r = (int)(t / d);
  out[t] += agg[t] / fmaxf(deg[r], 1.f);
}

__global__ void k_concat(const float* __restrict__ a, const float* __restrict__ x,
                         float* __restrict__ out, int n) {
  long t = (long)blockIdx.x * blockDim.x + threadIdx.x;
  if (t >= (long)n * (SPROJ + IN_CH)) return;
  int r = (int)(t / (SPROJ + IN_CH)), c = (int)(t % (SPROJ + IN_CH));
  out[t] = (c < SPROJ) ? a[(long)r * SPROJ + c] : x[(long)r * IN_CH + (c - SPROJ)];
}

// regular_nonlin: sample circular signal at NNL angles, ReLU, project back
__global__ void k_nonlin(const float* __restrict__ in, float* __restrict__ out,
                         int n, int mult, int L) {
  int t = blockIdx.x * blockDim.x + threadIdx.x;
  if (t >= n * mult) return;
  int node = t / mult, m = t - node * mult;
  int d = mult * (1 + 2 * L);
  long base = (long)node * d;
  float s = in[base + m];
  float av[3], bv[3];
  for (int l = 0; l < L; ++l) {
    av[l] = in[base + mult + l * 2 * mult + 2 * m];
    bv[l] = in[base + mult + l * 2 * mult + 2 * m + 1];
  }
  float g[NNL];
  float s2 = 0.f;
  for (int k = 0; k < NNL; ++k) {
    float th = (PI2f / NNL) * (float)k;
    float f = s;
    for (int l = 0; l < L; ++l) {
      float so, co; __sincosf((float)(l + 1) * th, &so, &co);
      f += av[l] * co + bv[l] * so;
    }
    g[k] = fmaxf(f, 0.f);
    s2 += g[k];
  }
  out[base + m] = s2 / (float)NNL;
  for (int l = 0; l < L; ++l) {
    float a2 = 0.f, b2 = 0.f;
    for (int k = 0; k < NNL; ++k) {
      float th = (PI2f / NNL) * (float)k;
      float so, co; __sincosf((float)(l + 1) * th, &so, &co);
      a2 += g[k] * co; b2 += g[k] * so;
    }
    out[base + mult + l * 2 * mult + 2 * m]     = a2 * (2.f / NNL);
    out[base + mult + l * 2 * mult + 2 * m + 1] = b2 * (2.f / NNL);
  }
}

// mean-pool [n,64] then MLP 64 -> 64(relu) -> 1, single block of 64 threads
__global__ void k_pool_head(const float* __restrict__ z, int n,
                            const float* __restrict__ W1, const float* __restrict__ b1,
                            const float* __restrict__ W2, const float* __restrict__ b2,
                            float* __restrict__ outp) {
  __shared__ float pooled[SPROJ];
  __shared__ float hidden[SPROJ];
  int c = threadIdx.x;
  float s = 0.f;
  for (int r = 0; r < n; ++r) s += z[(long)r * SPROJ + c];
  pooled[c] = s / (float)n;
  __syncthreads();
  float a = b1[c];
  for (int j = 0; j < SPROJ; ++j) a += pooled[j] * W1[j * SPROJ + c];
  hidden[c] = fmaxf(a, 0.f);
  __syncthreads();
  if (c == 0) {
    float o = b2[0];
    for (int j = 0; j < SPROJ; ++j) o += hidden[j] * W2[j];
    *outp = o;
  }
}

__global__ void k_cp_final(const float* __restrict__ hid, const float* __restrict__ W2,
                           const float* __restrict__ b2, float* __restrict__ out, int n) {
  int t = blockIdx.x * blockDim.x + threadIdx.x;
  if (t >= n) return;
  float a = b2[0];
  for (int j = 0; j < HEAD_H; ++j) a += hid[(long)t * HEAD_H + j] * W2[j];
  out[t] = a;
}

// wss edge conv (din=96, dout=2): VALU, per-edge thread, atomic scatter
__global__ void k_wss_edge(const float* __restrict__ hf,
                           const int* __restrict__ src, const int* __restrict__ dst,
                           const float* __restrict__ ang, const float* __restrict__ trn,
                           const float* __restrict__ W0, const float* __restrict__ Wc,
                           const float* __restrict__ Ws,
                           float* __restrict__ agg2, int E) {
  int e = blockIdx.x * blockDim.x + threadIdx.x;
  if (e >= E) return;
  int sn = src[e];
  float sa, ca; __sincosf(ang[e], &sa, &ca);
  float st, ct; __sincosf(trn[e], &st, &ct);
  const float* hr = hf + (long)sn * 96;
  float m0 = 0.f, m1 = 0.f;
  for (int j = 0; j < 32; ++j) {
    float v = hr[j];
    m0 += v * (W0[j * 2]     + ca * Wc[j * 2]     + sa * Ws[j * 2]);
    m1 += v * (W0[j * 2 + 1] + ca * Wc[j * 2 + 1] + sa * Ws[j * 2 + 1]);
  }
  for (int p = 0; p < 32; ++p) {
    float a = hr[32 + 2 * p], b = hr[32 + 2 * p + 1];
    float ra = a * ct - b * st;
    float rb = a * st + b * ct;
    int ja = 32 + 2 * p, jb = ja + 1;
    m0 += ra * (W0[ja * 2]     + ca * Wc[ja * 2]     + sa * Ws[ja * 2]);
    m1 += ra * (W0[ja * 2 + 1] + ca * Wc[ja * 2 + 1] + sa * Ws[ja * 2 + 1]);
    m0 += rb * (W0[jb * 2]     + ca * Wc[jb * 2]     + sa * Ws[jb * 2]);
    m1 += rb * (W0[jb * 2 + 1] + ca * Wc[jb * 2 + 1] + sa * Ws[jb * 2 + 1]);
  }
  int dn = dst[e];
  atomicAdd(&agg2[(long)dn * 2],     m0);
  atomicAdd(&agg2[(long)dn * 2 + 1], m1);
}

__global__ void k_wss_final(const float* __restrict__ hf, const float* __restrict__ Wself,
                            const float* __restrict__ agg2, const float* __restrict__ deg,
                            const float* __restrict__ e1, const float* __restrict__ e2,
                            float* __restrict__ out, int n) {
  int t = blockIdx.x * blockDim.x + threadIdx.x;
  if (t >= n) return;
  float dg = fmaxf(deg[t], 1.f);
  float v0 = agg2[(long)t * 2] / dg, v1 = agg2[(long)t * 2 + 1] / dg;
  const float* hr = hf + (long)t * 96;
  for (int j = 0; j < 96; ++j) {
    float f = hr[j];
    v0 += f * Wself[j * 2];
    v1 += f * Wself[j * 2 + 1];
  }
  for (int k = 0; k < 3; ++k)
    out[(long)t * 3 + k] = v0 * e1[(long)t * 3 + k] + v1 * e2[(long)t * 3 + k];
}

// ---------------- B-operand packing (WMMA fragment layout) ----------------
// Wp layout: [N/16][K32][lane 0..31][e 0..15] bf16, zero-padded K chunks.
// Fragment mapping: lane n holds col = ntile*16 + (n&15), K = (e&7) + 16*(e>=8) + 8*(n>=16).

__global__ void k_pack_w(const float* __restrict__ W, bf16* __restrict__ Wp,
                         int K, int N, int K32) {
  long total = (long)(N >> 4) * K32 * 512;
  long t = (long)blockIdx.x * blockDim.x + threadIdx.x;
  if (t >= total) return;
  int e = (int)(t & 15);
  int lane = (int)((t >> 4) & 31);
  long rest = t >> 9;
  int kt = (int)(rest % K32);
  int nt = (int)(rest / K32);
  int kb = kt * 32 + (e & 7) + ((e >= 8) ? 16 : 0) + ((lane >= 16) ? 8 : 0);
  int col = nt * 16 + (lane & 15);
  Wp[t] = f2bf((kb < K) ? W[(long)kb * N + col] : 0.f);
}

// stacked angle-kernel weights: rows [0,din)=W0, [din,2din)=Wc, [2din,3din)=Ws
__global__ void k_pack_stack3(const float* __restrict__ W0, const float* __restrict__ Wc,
                              const float* __restrict__ Ws, bf16* __restrict__ Wp,
                              int din, int dout) {
  int Ktot = 3 * din;
  int K32 = ((Ktot + 63) & ~63) >> 5;
  long total = (long)(dout >> 4) * K32 * 512;
  long t = (long)blockIdx.x * blockDim.x + threadIdx.x;
  if (t >= total) return;
  int e = (int)(t & 15);
  int lane = (int)((t >> 4) & 31);
  long rest = t >> 9;
  int kt = (int)(rest % K32);
  int nt = (int)(rest / K32);
  int kb = kt * 32 + (e & 7) + ((e >= 8) ? 16 : 0) + ((lane >= 16) ? 8 : 0);
  int col = nt * 16 + (lane & 15);
  float v = 0.f;
  if (kb < Ktot) {
    int part = kb / din, row = kb - part * din;
    const float* src = (part == 0) ? W0 : ((part == 1) ? Wc : Ws);
    v = src[(long)row * dout + col];
  }
  Wp[t] = f2bf(v);
}

// ---------------- WMMA GEMM ----------------
// C[M,N] = A[M,K] @ B[K,N] (+bias)(+relu); A fp32 row-major, B pre-packed bf16.
// M, N multiples of 16. NT (1..4) = compile-time n-tiles per wave: one A
// fragment per 32-K chunk feeds NT WMMAs on independent accumulators
// (single-pass A streaming; B streams are L2-resident packed bf16).
// NTA: non-temporal A loads for one-shot streams (interp matrix).
// grid (nFullBlocks or 1, ceil(M/64)), block (32,4).

template <int NT, bool NTA>
__global__ void k_gemm_wmma_t(const float* __restrict__ A, const bf16* __restrict__ Bp,
                              const float* __restrict__ bias, float* __restrict__ C,
                              int M, int N, int K, int relu, int ntBase) {
  int lane = threadIdx.x;
  int hi16 = (lane >= 16) ? 1 : 0;
  int l15 = lane & 15;
  int nt0 = ntBase + blockIdx.x * 4;
  int m0 = (blockIdx.y * blockDim.y + threadIdx.y) * 16;
  if (m0 >= M) return;                       // wave-uniform exit
  int K32 = (K + 31) >> 5;
  size_t bstride = (size_t)K32 * 512;
  const bf16* bp0 = Bp + (size_t)nt0 * bstride + (size_t)lane * 16;
  const bf16* bp1 = bp0 + bstride;
  const bf16* bp2 = bp1 + bstride;
  const bf16* bp3 = bp2 + bstride;
  const float* arowB = A + (size_t)(m0 + l15) * K;   // row base
  const float* arow = arowB + hi16 * 8;              // folded lane K offset
  v8f acc0 = {}, acc1 = {}, acc2 = {}, acc3 = {};
  int k0 = 0;
  for (; k0 + 32 <= K; k0 += 32) {           // guard-free straight-line loop
    v16bf a;
    put4(a, 0,  ld4<NTA>(arow + k0));
    put4(a, 4,  ld4<NTA>(arow + k0 + 4));
    put4(a, 8,  ld4<NTA>(arow + k0 + 16));
    put4(a, 12, ld4<NTA>(arow + k0 + 20));
    { v16bf b = *(const v16bf*)(bp0); bp0 += 512; acc0 = wmma_bf16(a, b, acc0); }
    if (NT > 1) { v16bf b = *(const v16bf*)(bp1); bp1 += 512; acc1 = wmma_bf16(a, b, acc1); }
    if (NT > 2) { v16bf b = *(const v16bf*)(bp2); bp2 += 512; acc2 = wmma_bf16(a, b, acc2); }
    if (NT > 3) { v16bf b = *(const v16bf*)(bp3); bp3 += 512; acc3 = wmma_bf16(a, b, acc3); }
  }
  if (k0 < K) {                               // branch-free tail: clamp + select
    v16bf a;
#pragma unroll
    for (int e = 0; e < 16; ++e) {
      int ka = k0 + (e & 7) + ((e >= 8) ? 16 : 0) + hi16 * 8;
      int kc = (ka < K) ? ka : (K - 1);
      float v = arowB[kc];
      a[e] = f2bf((ka < K) ? v : 0.f);
    }
    { v16bf b = *(const v16bf*)(bp0); acc0 = wmma_bf16(a, b, acc0); }   // B is K-padded
    if (NT > 1) { v16bf b = *(const v16bf*)(bp1); acc1 = wmma_bf16(a, b, acc1); }
    if (NT > 2) { v16bf b = *(const v16bf*)(bp2); acc2 = wmma_bf16(a, b, acc2); }
    if (NT > 3) { v16bf b = *(const v16bf*)(bp3); acc3 = wmma_bf16(a, b, acc3); }
  }
  store_tile(C, acc0, bias, m0, (nt0 + 0) * 16 + l15, N, hi16, relu);
  if (NT > 1) store_tile(C, acc1, bias, m0, (nt0 + 1) * 16 + l15, N, hi16, relu);
  if (NT > 2) store_tile(C, acc2, bias, m0, (nt0 + 2) * 16 + l15, N, hi16, relu);
  if (NT > 3) store_tile(C, acc3, bias, m0, (nt0 + 3) * 16 + l15, N, hi16, relu);
}

// ---------------- fused gauge-equivariant edge conv ----------------
// Per 16-edge tile: gather h[src], rotate irreps by transporter, stage
// [msg | msg*cos(ang) | msg*sin(ang)] bf16 in LDS (16 x Kpad, Kpad = 3*din
// rounded to 64, zero-padded), then WMMA against packed Wcat[3*din, dout]
// and atomic-scatter rows into agg[dst] (segment-sum).
// block = 128 (4 waves); grid = E/16; dyn LDS = 16*Kpad*2 bytes.

__global__ void k_gem_conv_wmma(const float* __restrict__ h,
                                const int* __restrict__ src, const int* __restrict__ dst,
                                const float* __restrict__ ang, const float* __restrict__ trn,
                                const bf16* __restrict__ Wcat,
                                float* __restrict__ agg,
                                int E, int din, int dout, int mult, int L) {
  extern __shared__ bf16 smem[];             // [16][Kpad]
  __shared__ int   eSrc[16];
  __shared__ float eCA[16], eSA[16], eCT[48], eST[48];
  const int Ktot = 3 * din;
  const int Kpad = (Ktot + 63) & ~63;
  const int e0 = blockIdx.x * 16;
  (void)E;
  // per-edge trig + src gather, once
  if (threadIdx.x < 16) {
    int edge = e0 + threadIdx.x;
    eSrc[threadIdx.x] = src[edge];
    float sa, ca; __sincosf(ang[edge], &sa, &ca);
    eCA[threadIdx.x] = ca; eSA[threadIdx.x] = sa;
    float tr = trn[edge];
    for (int o = 1; o <= L; ++o) {
      float so, co; __sincosf((float)o * tr, &so, &co);
      eCT[(o - 1) * 16 + threadIdx.x] = co;
      eST[(o - 1) * 16 + threadIdx.x] = so;
    }
  }
  __syncthreads();
  // stage 1: rotate + angle-scale into LDS
  for (int idx = threadIdx.x; idx < 16 * Ktot; idx += blockDim.x) {
    int er = idx / Ktot;
    int j = idx - er * Ktot;
    int part = j / din;
    int jj = j - part * din;
    const float* hr = h + (long)eSrc[er] * din;
    float v;
    if (jj < mult) {
      v = hr[jj];
    } else {
      int q = jj - mult;
      int o = q / (2 * mult);                // order-1, 0-based
      int r = q - o * 2 * mult;
      int p = r >> 1, isb = r & 1;
      float a = hr[mult + o * 2 * mult + 2 * p];
      float b = hr[mult + o * 2 * mult + 2 * p + 1];
      float co = eCT[o * 16 + er], so = eST[o * 16 + er];
      v = isb ? (a * so + b * co) : (a * co - b * so);
    }
    if (part == 1) v *= eCA[er];
    else if (part == 2) v *= eSA[er];
    smem[er * Kpad + j] = f2bf(v);
  }
  if (Kpad > Ktot) {                         // zero K padding
    int padw = Kpad - Ktot;
    for (int idx = threadIdx.x; idx < 16 * padw; idx += blockDim.x) {
      int er = idx / padw;
      int j = Ktot + (idx - er * padw);
      smem[er * Kpad + j] = f2bf(0.f);
    }
  }
  __syncthreads();
  // stage 2: WMMA over Kpad (guard-free), atomic scatter
  int wave = threadIdx.x >> 5;
  int lane = threadIdx.x & 31;
  int hi16 = (lane >= 16) ? 1 : 0;
  int l15 = lane & 15;
  int nT = dout >> 4;
  int nWaves = blockDim.x >> 5;
  int K32 = Kpad >> 5;
  const bf16* arow = smem + l15 * Kpad + hi16 * 8;
  for (int nt = wave; nt < nT; nt += nWaves) {
    const bf16* bp = Wcat + (size_t)nt * K32 * 512 + (size_t)lane * 16;
    v8f acc0 = {}, acc1 = {};
    for (int k0 = 0; k0 < Kpad; k0 += 64) {
      v8bf lo0 = *(const v8bf*)(arow + k0);
      v8bf hi0 = *(const v8bf*)(arow + k0 + 16);
      v8bf lo1 = *(const v8bf*)(arow + k0 + 32);
      v8bf hi1 = *(const v8bf*)(arow + k0 + 48);
      v16bf a0 = __builtin_shufflevector(lo0, hi0, 0,1,2,3,4,5,6,7,8,9,10,11,12,13,14,15);
      v16bf a1 = __builtin_shufflevector(lo1, hi1, 0,1,2,3,4,5,6,7,8,9,10,11,12,13,14,15);
      v16bf b0 = *(const v16bf*)(bp);
      v16bf b1 = *(const v16bf*)(bp + 512);
      bp += 1024;
      acc0 = wmma_bf16(a0, b0, acc0);
      acc1 = wmma_bf16(a1, b1, acc1);
    }
    v8f acc = acc0 + acc1;
#pragma unroll
    for (int r = 0; r < 8; ++r) {
      int row = r + hi16 * 8;
      int dn = dst[e0 + row];
      atomicAdd(&agg[(long)dn * dout + nt * 16 + l15], acc[r]);
    }
  }
}

// ---------------- host ----------------

extern "C" void kernel_launch(void* const* d_in, const int* in_sizes, int n_in,
                              void* d_out, int out_size, void* d_ws, size_t ws_size,
                              hipStream_t stream) {
  (void)in_sizes; (void)n_in; (void)out_size; (void)ws_size;
  const float* x    = (const float*)d_in[0];
  const int*   fei  = (const int*)d_in[1];
  const int*   fsrc = fei;
  const int*   fdst = fei + E_FINE;
  const float* fang = (const float*)d_in[2];
  const float* ftrn = (const float*)d_in[3];
  const int*   cidx = (const int*)d_in[4];
  const int*   cei  = (const int*)d_in[5];
  const int*   csrc = cei;
  const int*   cdst = cei + E_COARSE;
  const float* cang = (const float*)d_in[6];
  const float* ctrn = (const float*)d_in[7];
  const float* interp = (const float*)d_in[8];
  const float* e1 = (const float*)d_in[9];
  const float* e2 = (const float*)d_in[10];
  // d_in[11] batch, d_in[12] coarse_batch: single graph, unused

  int pi = 13;
  auto P = [&]() -> const float* { return (const float*)d_in[pi++]; };
  const float* embW = P(); const float* embB = P();
  const float *cWself[6], *cW0[6], *cWc[6], *cWs[6], *cBias[6];
  for (int i = 0; i < 6; ++i) { cWself[i]=P(); cW0[i]=P(); cWc[i]=P(); cWs[i]=P(); cBias[i]=P(); }
  const float* symW = P(); const float* symB = P();
  const float* cdW1 = P(); const float* cdB1 = P(); const float* cdW2 = P(); const float* cdB2 = P();
  const float* clW1 = P(); const float* clB1 = P(); const float* clW2 = P(); const float* clB2 = P();
  const float* upWproj = P(); const float* upBproj = P();
  const float* upWout  = P(); const float* upBout  = P();
  const float *rWself[2], *rW0[2], *rWc[2], *rWs[2], *rBias[2];
  for (int i = 0; i < 2; ++i) { rWself[i]=P(); rW0[i]=P(); rWc[i]=P(); rWs[i]=P(); rBias[i]=P(); }
  const float* cpSymW = P(); const float* cpSymB = P();
  const float* cpW1 = P(); const float* cpB1 = P(); const float* cpW2 = P(); const float* cpB2 = P();
  const float* wssWself = P(); const float* wssW0 = P(); const float* wssWc = P(); const float* wssWs = P();

  // workspace carve-out
  char* ws = (char*)d_ws;
  size_t cur = 0;
  auto alloc = [&](size_t bytes) -> void* {
    void* p = ws + cur;
    cur = (cur + bytes + 255) & ~(size_t)255;
    return p;
  };
  float* hF    = (float*)alloc((size_t)N_FINE * 80 * 4);
  float* cA    = (float*)alloc((size_t)N_COARSE * 448 * 4);
  float* cT    = (float*)alloc((size_t)N_COARSE * 448 * 4);
  float* aggC  = (float*)alloc((size_t)N_COARSE * 448 * 4);
  float* degC  = (float*)alloc((size_t)N_COARSE * 4);
  bf16*  Wcat  = (bf16*) alloc((size_t)1344 * 448 * 2);        // packed conv weights
  bf16*  Wpack = (bf16*) alloc((size_t)4000 * 64 * 2);         // packed GEMM B (max: proj)
  float* hcs   = (float*)alloc((size_t)N_COARSE * 64 * 4);
  float* proj  = (float*)alloc((size_t)N_COARSE * 64 * 4);
  float* itp   = (float*)alloc((size_t)N_FINE * 64 * 4);
  float* ccat  = (float*)alloc((size_t)N_FINE * 68 * 4);
  float* fA    = (float*)alloc((size_t)N_FINE * 96 * 4);
  float* fT    = (float*)alloc((size_t)N_FINE * 96 * 4);
  float* aggF  = (float*)alloc((size_t)N_FINE * 96 * 4);
  float* degF  = (float*)alloc((size_t)N_FINE * 4);
  float* hcp   = (float*)alloc((size_t)N_FINE * 64 * 4);
  float* hid   = (float*)alloc((size_t)N_FINE * HEAD_H * 4);
  float* agg2  = (float*)alloc((size_t)N_FINE * 2 * 4);

  float* out = (float*)d_out;   // [cp:40000][wss:120000][cd:1][cl:1]

  auto g1 = [](long n, int b) { return dim3((unsigned)((n + b - 1) / b)); };
  auto gemm = [&](const float* A, const float* W, const float* bias, float* C,
                  int M, int N, int K, int relu, bool ntA) {
    int K32 = (K + 31) >> 5;
    long total = (long)(N >> 4) * K32 * 512;
    k_pack_w<<<g1(total, 256), 256, 0, stream>>>(W, Wpack, K, N, K32);
    int N16 = N >> 4;
    int full = N16 >> 2;
    int rem = N16 & 3;
    dim3 bb(32, 4);
    unsigned gy = (unsigned)((M + 63) >> 6);
    if (full > 0) {
      if (ntA) k_gemm_wmma_t<4, true ><<<dim3(full, gy), bb, 0, stream>>>(A, Wpack, bias, C, M, N, K, relu, 0);
      else     k_gemm_wmma_t<4, false><<<dim3(full, gy), bb, 0, stream>>>(A, Wpack, bias, C, M, N, K, relu, 0);
    }
    if (rem == 1)      k_gemm_wmma_t<1, false><<<dim3(1, gy), bb, 0, stream>>>(A, Wpack, bias, C, M, N, K, relu, full * 4);
    else if (rem == 2) k_gemm_wmma_t<2, false><<<dim3(1, gy), bb, 0, stream>>>(A, Wpack, bias, C, M, N, K, relu, full * 4);
    else if (rem == 3) k_gemm_wmma_t<3, false><<<dim3(1, gy), bb, 0, stream>>>(A, Wpack, bias, C, M, N, K, relu, full * 4);
  };

  const int multO[6] = {16, 32, 32, 64, 64, 64};
  const int LO[6]    = {2, 2, 3, 3, 2, 1};
  const int multI[6] = {16, 16, 32, 32, 64, 64};
  const int LI[6]    = {2, 2, 2, 3, 3, 2};
  const int dims[7]  = {80, 80, 160, 224, 448, 320, 192};

  // input embed + gather coarse nodes
  k_linear_small<<<g1((long)N_FINE * 80, 256), 256, 0, stream>>>(x, embW, embB, hF, N_FINE, IN_CH, 80);
  k_gather_rows<<<g1((long)N_COARSE * 80, 256), 256, 0, stream>>>(hF, cidx, cA, N_COARSE, 80);

  // coarse degrees (same for all layers)
  (void)hipMemsetAsync(degC, 0, (size_t)N_COARSE * 4, stream);
  k_edge_deg<<<g1(E_COARSE, 256), 256, 0, stream>>>(cdst, degC, E_COARSE);

  // coarse GEM conv stack
  for (int i = 0; i < 6; ++i) {
    int din = dims[i], dout = dims[i + 1];
    int Ktot = 3 * din, Kpad = (Ktot + 63) & ~63;
    (void)hipMemsetAsync(aggC, 0, (size_t)N_COARSE * dout * 4, stream);
    k_pack_stack3<<<g1((long)(dout >> 4) * (Kpad >> 5) * 512, 256), 256, 0, stream>>>(
        cW0[i], cWc[i], cWs[i], Wcat, din, dout);
    k_gem_conv_wmma<<<E_COARSE / 16, 128, (size_t)16 * Kpad * sizeof(bf16), stream>>>(
        cA, csrc, cdst, cang, ctrn, Wcat, aggC, E_COARSE, din, dout, multI[i], LI[i]);
    gemm(cA, cWself[i], cBias[i], cT, N_COARSE, dout, din, 0, false);   // h@Wself + b
    k_finalize_conv<<<g1((long)N_COARSE * dout, 256), 256, 0, stream>>>(cT, aggC, degC, N_COARSE, dout);
    k_nonlin<<<g1((long)N_COARSE * multO[i], 256), 256, 0, stream>>>(cT, cA, N_COARSE, multO[i], LO[i]);
  }

  // symmetry-break + pooled heads
  gemm(cA, symW, symB, hcs, N_COARSE, SPROJ, 192, 0, false);
  k_pool_head<<<1, SPROJ, 0, stream>>>(hcs, N_COARSE, cdW1, cdB1, cdW2, cdB2, out + 160000);
  k_pool_head<<<1, SPROJ, 0, stream>>>(hcs, N_COARSE, clW1, clB1, clW2, clB2, out + 160001);

  // unpool: proj -> dense interp GEMM (single-pass 640MB NT stream) -> concat -> project
  gemm(cA, upWproj, upBproj, proj, N_COARSE, SPROJ, 192, 0, false);
  gemm(interp, proj, nullptr, itp, N_FINE, SPROJ, N_COARSE, 0, true);
  k_concat<<<g1((long)N_FINE * 68, 256), 256, 0, stream>>>(itp, x, ccat, N_FINE);
  gemm(ccat, upWout, upBout, fA, N_FINE, 96, SPROJ + IN_CH, 0, false);

  // fine degrees
  (void)hipMemsetAsync(degF, 0, (size_t)N_FINE * 4, stream);
  k_edge_deg<<<g1(E_FINE, 256), 256, 0, stream>>>(fdst, degF, E_FINE);

  // refine GEM conv stack (din=dout=96, ftype (32,1))
  for (int i = 0; i < 2; ++i) {
    int Kpad = (288 + 63) & ~63;   // 320
    (void)hipMemsetAsync(aggF, 0, (size_t)N_FINE * 96 * 4, stream);
    k_pack_stack3<<<g1((long)(96 >> 4) * (Kpad >> 5) * 512, 256), 256, 0, stream>>>(
        rW0[i], rWc[i], rWs[i], Wcat, 96, 96);
    k_gem_conv_wmma<<<E_FINE / 16, 128, (size_t)16 * Kpad * sizeof(bf16), stream>>>(
        fA, fsrc, fdst, fang, ftrn, Wcat, aggF, E_FINE, 96, 96, 32, 1);
    gemm(fA, rWself[i], rBias[i], fT, N_FINE, 96, 96, 0, false);
    k_finalize_conv<<<g1((long)N_FINE * 96, 256), 256, 0, stream>>>(fT, aggF, degF, N_FINE, 96);
    k_nonlin<<<g1((long)N_FINE * 32, 256), 256, 0, stream>>>(fT, fA, N_FINE, 32, 1);
  }

  // cp head
  gemm(fA, cpSymW, cpSymB, hcp, N_FINE, SPROJ, 96, 0, false);
  gemm(hcp, cpW1, cpB1, hid, N_FINE, HEAD_H, SPROJ, 1, false);
  k_cp_final<<<g1(N_FINE, 256), 256, 0, stream>>>(hid, cpW2, cpB2, out, N_FINE);

  // wss head (dout=2 -> VALU)
  (void)hipMemsetAsync(agg2, 0, (size_t)N_FINE * 2 * 4, stream);
  k_wss_edge<<<g1(E_FINE, 256), 256, 0, stream>>>(fA, fsrc, fdst, fang, ftrn, wssW0, wssWc, wssWs, agg2, E_FINE);
  k_wss_final<<<g1(N_FINE, 256), 256, 0, stream>>>(fA, wssWself, agg2, degF, e1, e2, out + N_FINE, N_FINE);
}
